// CNN2_11570641895609
// MI455X (gfx1250) — matrix operand order repaired and checked by reading, not compiled, not run on hardware
//
#include <hip/hip_runtime.h>
#include <hip/hip_bf16.h>

typedef __attribute__((ext_vector_type(16))) __bf16 v16bf;
typedef __attribute__((ext_vector_type(8)))  __bf16 v8bf;
typedef __attribute__((ext_vector_type(8)))  float  v8f;

#define SUBJ   8
#define R      116
#define T      256
#define RP     128      // padded ROI dim (8 WMMA tiles)
#define QCOLS  768      // 256 zero | 256 data | 256 zero
#define NLAGS  511
#define LAGS_PER_WG 16
#define NCHUNK 32       // 32*16 = 512 >= 511

// workspace layout (bytes)
#define WS_QPAD   0                         // 8*128*768*2  = 1,572,864
#define WS_ACCUM  1572864                   // 8*2*128*128*4 = 1,048,576
#define WS_GBUF   2621440                   // 8*2*116*116*4 =   861,184
#define WS_P1     3482624                   // 8*8*14*14*4   =    50,176
#define WS_P2     3532800                   // 8*16*3*3*4    =     4,608

// ---------------------------------------------------------------- zero fill
__global__ void zero_f32(float* __restrict__ p, int n) {
  int i = blockIdx.x * blockDim.x + threadIdx.x;
  int stride = gridDim.x * blockDim.x;
  for (; i < n; i += stride) p[i] = 0.0f;
}

// ------------------------------------------- prep: y = W @ Y  -> bf16 Qpad
__global__ void prep_kernel(const float* __restrict__ Y,
                            const float* __restrict__ W,
                            __bf16* __restrict__ Qp) {
  const int i = blockIdx.x;    // 0..115
  const int b = blockIdx.y;    // subject
  const int t = threadIdx.x;   // 0..255
  const float* wrow = W + i * R;
  const float* yb   = Y + (size_t)(b * R) * T + t;
  float acc = 0.f;
  for (int j = 0; j < R; ++j) acc += wrow[j] * yb[(size_t)j * T];
  Qp[(size_t)(b * RP + i) * QCOLS + 256 + t] = (__bf16)acc;
}

// --------------------------------------------- main cross-correlation GEMM
union AFrag { v16bf v; v8bf h[2]; };

__launch_bounds__(256)
__global__ void corr_kernel(const __bf16* __restrict__ Qg,
                            float* __restrict__ accum) {
  extern __shared__ char smem[];
  __bf16* Q    = (__bf16*)smem;                       // 128 x 768 bf16 (192 KB)
  float* trLDS = (float*)(smem + RP * QCOLS * sizeof(__bf16));

  const int tid   = threadIdx.x;
  const int wave  = tid >> 5;
  const int lane  = tid & 31;
  const int mg    = wave >> 1;   // 0..3 -> owns mtiles {2mg, 2mg+1}
  const int ng    = wave & 1;    // 0..1 -> owns ntiles {4ng .. 4ng+3}
  const int chunk = blockIdx.x;
  const int subj  = blockIdx.y;

  // async copy of this subject's padded signal, global -> LDS (CDNA5 path)
  {
    const uint4* src = (const uint4*)(Qg + (size_t)subj * RP * QCOLS);
    const unsigned ldsbase = (unsigned)(uintptr_t)Q;  // low 32 bits = LDS offset
    const int n16 = RP * QCOLS * 2 / 16;              // 12288 16B chunks
    for (int idx = tid; idx < n16; idx += 256) {
      unsigned loff = ldsbase + (unsigned)idx * 16u;
      const uint4* g = src + idx;
      asm volatile("global_load_async_to_lds_b128 %0, %1, off"
                   :: "v"(loff), "v"(g) : "memory");
    }
    asm volatile("s_wait_asynccnt 0x0" ::: "memory");
  }
  __syncthreads();

  // cache all A fragments for this wave's 2 row-tiles (lag-invariant)
  const int arow  = lane & 15;
  const int akoff = (lane >> 4) << 3;
  AFrag Ac[2][8];
  #pragma unroll
  for (int mi = 0; mi < 2; ++mi) {
    const __bf16* abase =
        Q + (size_t)((2 * mg + mi) * 16 + arow) * QCOLS + 256 + akoff;
    #pragma unroll
    for (int kt = 0; kt < 8; ++kt) {
      Ac[mi][kt].h[0] = *(const v8bf*)(abase + kt * 32);
      Ac[mi][kt].h[1] = *(const v8bf*)(abase + kt * 32 + 16);
    }
  }

  float accS [2][4][8];
  float accS2[2][4][8];
  #pragma unroll
  for (int mi = 0; mi < 2; ++mi)
    #pragma unroll
    for (int ni = 0; ni < 4; ++ni)
      #pragma unroll
      for (int v = 0; v < 8; ++v) { accS[mi][ni][v] = 0.f; accS2[mi][ni][v] = 0.f; }

  const int bkoff = (lane >> 4) << 3;
  const __bf16* Brow[4];
  #pragma unroll
  for (int ni = 0; ni < 4; ++ni)
    Brow[ni] = Q + (size_t)((4 * ng + ni) * 16 + (lane & 15)) * QCOLS + bkoff;

  for (int il = 0; il < LAGS_PER_WG; ++il) {
    const int l = chunk * LAGS_PER_WG + il;
    if (l >= NLAGS) break;
    const int o2 = NLAGS - l;            // shifted-B start column, 1..511

    if (tid == 0) *trLDS = 0.f;
    __syncthreads();

    v8f C[2][4];
    #pragma unroll
    for (int mi = 0; mi < 2; ++mi)
      #pragma unroll
      for (int ni = 0; ni < 4; ++ni)
        C[mi][ni] = (v8f){0.f, 0.f, 0.f, 0.f, 0.f, 0.f, 0.f, 0.f};

    for (int kt = 0; kt < 8; ++kt) {
      #pragma unroll
      for (int ni = 0; ni < 4; ++ni) {
        v16bf Bv;
        const __bf16* bp = Brow[ni] + o2 + kt * 32;
        #pragma unroll
        for (int k = 0; k < 16; ++k) Bv[k] = bp[k + ((k >> 3) << 3)];
        #pragma unroll
        for (int mi = 0; mi < 2; ++mi)
          C[mi][ni] = __builtin_amdgcn_wmma_f32_16x16x32_bf16(
              false, Ac[mi][kt].v, false, Bv, (short)0, C[mi][ni], false, false);
      }
    }

    // trace: diagonal tiles are those with mtile == ntile; C layout:
    // lanes 0-15: (M=v, N=lane); lanes 16-31: (M=v+8, N=lane-16)
    float d = 0.f;
    #pragma unroll
    for (int mi = 0; mi < 2; ++mi)
      #pragma unroll
      for (int ni = 0; ni < 4; ++ni) {
        bool diag = (2 * mg + mi) == (4 * ng + ni);
        #pragma unroll
        for (int v = 0; v < 8; ++v) {
          bool isd = diag && ((lane == v) || (lane == v + 24));
          d += isd ? C[mi][ni][v] : 0.f;
        }
      }
    atomicAdd(trLDS, d);
    __syncthreads();

    const float inv = 1.0f / (*trLDS);
    #pragma unroll
    for (int mi = 0; mi < 2; ++mi)
      #pragma unroll
      for (int ni = 0; ni < 4; ++ni)
        #pragma unroll
        for (int v = 0; v < 8; ++v) {
          float s = C[mi][ni][v] * inv;
          accS [mi][ni][v] += s;
          accS2[mi][ni][v] += s * s;
        }
    __syncthreads();   // protect trLDS reset of next lag
  }

  // flush per-chunk partial sums with float atomics
  float* gS  = accum + (size_t)(subj * 2 + 0) * RP * RP;
  float* gS2 = accum + (size_t)(subj * 2 + 1) * RP * RP;
  const int rbase = (lane >> 4) << 3;
  #pragma unroll
  for (int mi = 0; mi < 2; ++mi)
    #pragma unroll
    for (int ni = 0; ni < 4; ++ni) {
      int col = (4 * ng + ni) * 16 + (lane & 15);
      #pragma unroll
      for (int v = 0; v < 8; ++v) {
        int row = (2 * mg + mi) * 16 + rbase + v;
        atomicAdd(&gS [row * RP + col], accS [mi][ni][v]);
        atomicAdd(&gS2[row * RP + col], accS2[mi][ni][v]);
      }
    }
}

// ------------------------------------------------- finalize: mean/var -> G
__global__ void finalize_kernel(const float* __restrict__ accum,
                                float* __restrict__ G) {
  const int i = blockIdx.x, b = blockIdx.y, j = threadIdx.x;
  if (j >= R) return;
  const float L = (float)NLAGS;
  float s  = accum[(size_t)(b * 2 + 0) * RP * RP + i * RP + j];
  float s2 = accum[(size_t)(b * 2 + 1) * RP * RP + i * RP + j];
  float mean = s / L;
  float var  = (s2 - s * s / L) / (L - 1.0f);
  G[((size_t)(b * 2 + 0) * R + i) * R + j] = 0.5f * (mean + 1.0f);
  G[((size_t)(b * 2 + 1) * R + i) * R + j] = 0.5f * (var  + 1.0f);
}

// --------------------------------------- conv1 (4x4, pad1) + leaky + pool8
__global__ void conv1_kernel(const float* __restrict__ G,
                             const float* __restrict__ w,
                             const float* __restrict__ bias,
                             float* __restrict__ P1) {
  const int blk = blockIdx.x;       // b*8 + oc
  const int b = blk >> 3, oc = blk & 7;
  const int tid = threadIdx.x;      // 0..195
  const int py = tid / 14, px = tid % 14;
  float m = -3.4e38f;
  for (int wy = 0; wy < 8; ++wy)
    for (int wx = 0; wx < 8; ++wx) {
      int oy = py * 8 + wy, ox = px * 8 + wx;   // conv output 115x115
      float acc = bias[oc];
      for (int ic = 0; ic < 2; ++ic)
        for (int ky = 0; ky < 4; ++ky) {
          int iy = oy + ky - 1;
          if (iy < 0 || iy >= R) continue;
          for (int kx = 0; kx < 4; ++kx) {
            int ix = ox + kx - 1;
            if (ix < 0 || ix >= R) continue;
            acc += w[((oc * 2 + ic) * 4 + ky) * 4 + kx] *
                   G[((size_t)(b * 2 + ic) * R + iy) * R + ix];
          }
        }
      float v = acc >= 0.f ? acc : 0.2f * acc;
      m = fmaxf(m, v);
    }
  P1[((b * 8 + oc) * 14 + py) * 14 + px] = m;
}

// --------------------------------------- conv2 (2x2, pad1) + leaky + pool4
__global__ void conv2_kernel(const float* __restrict__ P1,
                             const float* __restrict__ w,
                             const float* __restrict__ bias,
                             float* __restrict__ P2) {
  const int blk = blockIdx.x;       // b*16 + oc
  const int b = blk >> 4, oc = blk & 15;
  const int tid = threadIdx.x;
  if (tid >= 9) return;
  const int py = tid / 3, px = tid % 3;
  float m = -3.4e38f;
  for (int wy = 0; wy < 4; ++wy)
    for (int wx = 0; wx < 4; ++wx) {
      int oy = py * 4 + wy, ox = px * 4 + wx;   // conv output 15x15
      float acc = bias[oc];
      for (int ic = 0; ic < 8; ++ic)
        for (int ky = 0; ky < 2; ++ky) {
          int iy = oy + ky - 1;
          if (iy < 0 || iy >= 14) continue;
          for (int kx = 0; kx < 2; ++kx) {
            int ix = ox + kx - 1;
            if (ix < 0 || ix >= 14) continue;
            acc += w[((oc * 8 + ic) * 2 + ky) * 2 + kx] *
                   P1[((b * 8 + ic) * 14 + iy) * 14 + ix];
          }
        }
      float v = acc >= 0.f ? acc : 0.2f * acc;
      m = fmaxf(m, v);
    }
  P2[((b * 16 + oc) * 3 + py) * 3 + px] = m;
}

// ------------------------------------------------------------ dense output
__global__ void dense_kernel(const float* __restrict__ P2,
                             const float* __restrict__ ow,
                             const float* __restrict__ ob,
                             float* __restrict__ out) {
  const int tid = threadIdx.x;
  if (tid >= 16) return;
  const int o = tid & 1, b = tid >> 1;
  float acc = ob[o];
  for (int k = 0; k < 144; ++k) acc += ow[o * 144 + k] * P2[b * 144 + k];
  out[b * 2 + o] = acc;
}

// --------------------------------------------------------------- launcher
extern "C" void kernel_launch(void* const* d_in, const int* in_sizes, int n_in,
                              void* d_out, int out_size, void* d_ws, size_t ws_size,
                              hipStream_t stream) {
  (void)in_sizes; (void)n_in; (void)out_size; (void)ws_size;
  const float* Y   = (const float*)d_in[0];
  const float* W   = (const float*)d_in[1];
  const float* c1w = (const float*)d_in[2];
  const float* c1b = (const float*)d_in[3];
  const float* c2w = (const float*)d_in[4];
  const float* c2b = (const float*)d_in[5];
  const float* ow  = (const float*)d_in[6];
  const float* ob  = (const float*)d_in[7];

  char* ws = (char*)d_ws;
  __bf16* Qp   = (__bf16*)(ws + WS_QPAD);
  float* accum = (float*)(ws + WS_ACCUM);
  float* Gbuf  = (float*)(ws + WS_GBUF);
  float* P1    = (float*)(ws + WS_P1);
  float* P2    = (float*)(ws + WS_P2);

  // zero Qpad (bf16 region) + accumulators every call (deterministic)
  const int nzero = (WS_GBUF - WS_QPAD) / 4;
  zero_f32<<<2560, 256, 0, stream>>>((float*)ws, nzero);

  prep_kernel<<<dim3(R, SUBJ), T, 0, stream>>>(Y, W, Qp);

  const size_t smem = (size_t)RP * QCOLS * sizeof(__bf16) + 16;  // 196624 B
  corr_kernel<<<dim3(NCHUNK, SUBJ), 256, smem, stream>>>(Qp, accum);

  finalize_kernel<<<dim3(R, SUBJ), 128, 0, stream>>>(accum, Gbuf);
  conv1_kernel<<<SUBJ * 8, 196, 0, stream>>>(Gbuf, c1w, c1b, P1);
  conv2_kernel<<<SUBJ * 16, 16, 0, stream>>>(P1, c2w, c2b, P2);
  dense_kernel<<<1, 16, 0, stream>>>(P2, ow, ob, (float*)d_out);
}
